// DGCNN_52192442581526
// MI455X (gfx1250) — compile-verified, matrix operand order checked by reading, not run on hardware
//
#include <hip/hip_runtime.h>
#include <hip/hip_bf16.h>

typedef __attribute__((ext_vector_type(16))) _Float16 v16h;
typedef __attribute__((ext_vector_type(8)))  _Float16 v8h;
typedef __attribute__((ext_vector_type(8)))  float    v8f;

#define NPTS 8192
#define KNN  20
#define BIGD 1e10f

// ---------------------------------------------------------------------------
// sq[i] = sum_c x[i,c]^2   (templated D -> fully unrolled, b128 loads for D=64)
template <int D>
__global__ void sqnorm_kernel(const float* __restrict__ X,
                              float* __restrict__ sq, int N) {
    int i = blockIdx.x * 256 + threadIdx.x;
    if (i >= N) return;
    float s = 0.f;
    if constexpr (D % 4 == 0) {
        const float4* x4 = (const float4*)(X + (size_t)i * D);
#pragma unroll
        for (int c = 0; c < D / 4; ++c) {
            float4 v = x4[c];
            s += v.x * v.x + v.y * v.y + v.z * v.z + v.w * v.w;
        }
    } else {
#pragma unroll
        for (int c = 0; c < D; ++c) { float v = X[(size_t)i * D + c]; s += v * v; }
    }
    sq[i] = s;
}

// ---------------------------------------------------------------------------
// One workgroup (256 threads) per point i: full 8192-entry distance row in LDS
// (32 KB of the 320 KB WGP LDS), then 20 argmin+mask passes.
// dist = sq_i + sq_j - 2 x_i.x_j, self = BIG; neighbor order irrelevant (max-aggr).
template <int D>
__global__ void knn_kernel(const float* __restrict__ X,
                           const float* __restrict__ sq,
                           int* __restrict__ idx, int N) {
    __shared__ float sdist[NPTS];
    __shared__ float sxi[(D + 3) & ~3];
    __shared__ float rv[256];
    __shared__ int   ri[256];

    const int i   = blockIdx.x;
    const int tid = threadIdx.x;

    if (tid < D) sxi[tid] = X[(size_t)i * D + tid];
    __syncthreads();

    const float sqi = sq[i];
    for (int j = tid; j < N; j += 256) {
        float dot = 0.f;
        if constexpr (D % 4 == 0) {
            const float4* xj4 = (const float4*)(X + (size_t)j * D);
            const float4* xi4 = (const float4*)sxi;
#pragma unroll
            for (int c = 0; c < D / 4; ++c) {
                float4 a = xi4[c], b = xj4[c];
                dot += a.x * b.x + a.y * b.y + a.z * b.z + a.w * b.w;
            }
        } else {
            const float* xj = X + (size_t)j * D;
#pragma unroll
            for (int c = 0; c < D; ++c) dot += sxi[c] * xj[c];
        }
        float dv = sqi + sq[j] - 2.f * dot;
        if (j == i) dv = BIGD;
        sdist[j] = dv;
    }
    __syncthreads();

    for (int t = 0; t < KNN; ++t) {
        float best = BIGD; int bi = 0;
        for (int j = tid; j < N; j += 256) {
            float v = sdist[j];
            if (v < best) { best = v; bi = j; }
        }
        rv[tid] = best; ri[tid] = bi;
        __syncthreads();
        for (int s = 128; s > 0; s >>= 1) {
            if (tid < s && rv[tid + s] < rv[tid]) {
                rv[tid] = rv[tid + s]; ri[tid] = ri[tid + s];
            }
            __syncthreads();
        }
        if (tid == 0) {
            idx[i * KNN + t] = ri[0];
            sdist[ri[0]] = BIGD;
        }
        __syncthreads();
    }
}

// ---------------------------------------------------------------------------
// Wt[o*Kpad + k] = (o<O && k<K) ? W[k*O + o] : 0   (f32 -> f16, transpose+pad)
__global__ void conv_wt_kernel(const float* __restrict__ W,
                               _Float16* __restrict__ Wt,
                               int K, int O, int Kpad, int Opad) {
    int id = blockIdx.x * 256 + threadIdx.x;
    if (id >= Opad * Kpad) return;
    int o = id / Kpad, kk = id % Kpad;
    float v = (o < O && kk < K) ? W[kk * O + o] : 0.f;
    Wt[id] = (_Float16)v;
}

// ---------------------------------------------------------------------------
// EdgeConv feature gather: feat[n*20+j, :] = [x_i, x_j - x_i] padded to Kp.
// Coalesced: consecutive threads write consecutive columns.
__global__ void build_feat_kernel(const float* __restrict__ X, int d, int Kp,
                                  const int* __restrict__ idx,
                                  _Float16* __restrict__ feat, int M) {
    int id = blockIdx.x * 256 + threadIdx.x;
    if (id >= M * Kp) return;
    int r = id / Kp, c = id % Kp;
    int i  = r / KNN;
    int jj = idx[i * KNN + (r % KNN)];
    float v;
    if (c < d)           v = X[(size_t)i  * d + c];
    else if (c < 2 * d)  v = X[(size_t)jj * d + (c - d)] - X[(size_t)i * d + (c - d)];
    else                 v = 0.f;
    feat[id] = (_Float16)v;
}

// ---------------------------------------------------------------------------
// Fragment loader: 16-bit A/B layout for one lane is two contiguous 8-half runs
//   elems 0..7  -> columns kc +      8*half + [0..7]   (VGPRs 0-3)
//   elems 8..15 -> columns kc + 16 + 8*half + [0..7]   (VGPRs 4-7)
// => exactly two global_load_b128 per fragment.
__device__ inline v16h load_frag(const _Float16* __restrict__ row, int kc, int half) {
    v8h lo = *(const v8h*)(row + kc + 8 * half);
    v8h hi = *(const v8h*)(row + kc + 16 + 8 * half);
    v16h f;
#pragma unroll
    for (int t = 0; t < 8; ++t) { f[t] = lo[t]; f[8 + t] = hi[t]; }
    return f;
}

// ---------------------------------------------------------------------------
// Generic WMMA GEMM: out = act(A[M,K] @ W[K,O] + b), W transposed [Opad,K].
// block = 128 (4 waves); each wave computes NT adjacent 16x16 O-tiles for one
// M-tile, reusing the A fragment NT times. grid = (M/16, Opad/(64*NT)).
template <int NT>
__global__ __launch_bounds__(128)
void gemm_wmma(const _Float16* __restrict__ A,   // [M,K]
               const _Float16* __restrict__ Wt,  // [Opad,K]
               const float* __restrict__ bias,   // [Oreal]
               int K, int Oreal,
               _Float16* __restrict__ outH,      // [M,Oreal] or null
               float* __restrict__ outF,         // [M,Oreal] or null
               int doRelu) {
    const int lane = threadIdx.x & 31;
    const int wave = threadIdx.x >> 5;
    const int half = lane >> 4;
    const int l16  = lane & 15;

    const int tileO0 = (blockIdx.y * 4 + wave) * NT;
    const int r      = blockIdx.x * 16 + l16;

    const _Float16* arow = A + (size_t)r * K;
    const _Float16* wrow[NT];
#pragma unroll
    for (int u = 0; u < NT; ++u)
        wrow[u] = Wt + (size_t)((tileO0 + u) * 16 + l16) * K;

    v8f acc[NT] = {};
    for (int kc = 0; kc < K; kc += 32) {
        v16h a = load_frag(arow, kc, half);
#pragma unroll
        for (int u = 0; u < NT; ++u) {
            v16h b = load_frag(wrow[u], kc, half);
            acc[u] = __builtin_amdgcn_wmma_f32_16x16x32_f16(
                false, a, false, b, (short)0, acc[u], false, false);
        }
    }

#pragma unroll
    for (int u = 0; u < NT; ++u) {
        const int col = (tileO0 + u) * 16 + l16;
        if (col >= Oreal) continue;
        const float bv = bias[col];
#pragma unroll
        for (int v = 0; v < 8; ++v) {
            const int row = blockIdx.x * 16 + v + 8 * half;
            float val = acc[u][v] + bv;
            if (doRelu) val = fmaxf(val, 0.f);
            if (outH) outH[(size_t)row * Oreal + col] = (_Float16)val;
            if (outF) outF[(size_t)row * Oreal + col] = val;
        }
    }
}

// ---------------------------------------------------------------------------
// max over k=20 neighbors -> x_out (f32, next kNN/gather) + slice of Hcat (f16)
__global__ void maxpool_kernel(const _Float16* __restrict__ h, // [N,20,64]
                               float* __restrict__ xout,       // [N,64]
                               _Float16* __restrict__ Hcat,    // [N,192]
                               int colOff, int N) {
    int id = blockIdx.x * 256 + threadIdx.x;
    if (id >= N * 64) return;
    int n = id >> 6, o = id & 63;
    float m = 0.f;  // all candidates are post-ReLU (>= 0)
#pragma unroll 4
    for (int j = 0; j < KNN; ++j)
        m = fmaxf(m, (float)h[((size_t)n * KNN + j) * 64 + o]);
    xout[(size_t)n * 64 + o] = m;
    Hcat[(size_t)n * 192 + colOff + o] = (_Float16)m;
}

// ---------------------------------------------------------------------------
// In-place log_softmax over C=50 columns; one 64-thread block per row.
__global__ void lsm_kernel(float* __restrict__ out, int C) {
    __shared__ float buf[64];
    const int row = blockIdx.x;
    const int t   = threadIdx.x;
    float v = (t < C) ? out[(size_t)row * C + t] : -1e30f;
    buf[t] = v; __syncthreads();
    for (int s = 32; s > 0; s >>= 1) {
        if (t < s) buf[t] = fmaxf(buf[t], buf[t + s]);
        __syncthreads();
    }
    const float mx = buf[0]; __syncthreads();
    float e = (t < C) ? __expf(v - mx) : 0.f;
    buf[t] = e; __syncthreads();
    for (int s = 32; s > 0; s >>= 1) {
        if (t < s) buf[t] += buf[t + s];
        __syncthreads();
    }
    const float lse = __logf(buf[0]);
    if (t < C) out[(size_t)row * C + t] = v - mx - lse;
}

// ---------------------------------------------------------------------------
extern "C" void kernel_launch(void* const* d_in, const int* in_sizes, int n_in,
                              void* d_out, int out_size, void* d_ws, size_t ws_size,
                              hipStream_t stream) {
    (void)in_sizes; (void)n_in; (void)out_size; (void)ws_size;
    const float* x   = (const float*)d_in[0];
    const float* W1  = (const float*)d_in[1];  const float* b1  = (const float*)d_in[2];
    const float* W2  = (const float*)d_in[3];  const float* b2  = (const float*)d_in[4];
    const float* W3  = (const float*)d_in[5];  const float* b3  = (const float*)d_in[6];
    const float* Wl  = (const float*)d_in[7];  const float* bl  = (const float*)d_in[8];
    const float* Wm1 = (const float*)d_in[9];  const float* bm1 = (const float*)d_in[10];
    const float* Wm2 = (const float*)d_in[11]; const float* bm2 = (const float*)d_in[12];
    const float* Wo  = (const float*)d_in[13]; const float* bo  = (const float*)d_in[14];

    const int N = NPTS;
    const int M = N * KNN;        // 163840 edge rows, multiple of 16

    char* ws = (char*)d_ws;
    size_t off = 0;
    auto carve = [&](size_t bytes) -> void* {
        void* p = ws + off;
        off = (off + bytes + 255) & ~(size_t)255;
        return p;
    };
    float*    sq   = (float*)   carve((size_t)N * 4);
    int*      nidx = (int*)     carve((size_t)N * KNN * 4);
    _Float16* feat = (_Float16*)carve((size_t)M * 128 * 2);   // reused as act1
    _Float16* h16  = (_Float16*)carve((size_t)M * 64 * 2);
    float*    x1   = (float*)   carve((size_t)N * 64 * 4);
    float*    x2   = (float*)   carve((size_t)N * 64 * 4);
    float*    x3   = (float*)   carve((size_t)N * 64 * 4);
    _Float16* Hcat = (_Float16*)carve((size_t)N * 192 * 2);
    _Float16* act2 = (_Float16*)carve((size_t)N * 256 * 2);
    _Float16* act3 = (_Float16*)carve((size_t)N * 128 * 2);
    _Float16* W1t  = (_Float16*)carve((size_t)64 * 32 * 2);
    _Float16* W2t  = (_Float16*)carve((size_t)64 * 128 * 2);
    _Float16* W3t  = (_Float16*)carve((size_t)64 * 128 * 2);
    _Float16* Wlt  = (_Float16*)carve((size_t)1024 * 192 * 2);
    _Float16* Wm1t = (_Float16*)carve((size_t)256 * 1024 * 2);
    _Float16* Wm2t = (_Float16*)carve((size_t)128 * 256 * 2);
    _Float16* Wot  = (_Float16*)carve((size_t)64 * 128 * 2);
    _Float16* act1 = feat;   // feat dead once conv3 GEMM finishes; alias for [N,1024]
    float*    outF = (float*)d_out;

    auto cvt = [&](const float* W, _Float16* Wt, int K, int O, int Kp, int Op) {
        int tot = Kp * Op;
        conv_wt_kernel<<<(tot + 255) / 256, 256, 0, stream>>>(W, Wt, K, O, Kp, Op);
    };
    cvt(W1, W1t, 6, 64, 32, 64);
    cvt(W2, W2t, 128, 64, 128, 64);
    cvt(W3, W3t, 128, 64, 128, 64);
    cvt(Wl, Wlt, 192, 1024, 192, 1024);
    cvt(Wm1, Wm1t, 1024, 256, 1024, 256);
    cvt(Wm2, Wm2t, 256, 128, 256, 128);
    cvt(Wo, Wot, 128, 50, 128, 64);

    const int Mtiles = M / 16;    // 10240
    const int Mr     = N / 16;    // 512

    // --- EdgeConv 1: kNN on raw 3-D points; feature K padded 6 -> 32
    sqnorm_kernel<3><<<(N + 255) / 256, 256, 0, stream>>>(x, sq, N);
    knn_kernel<3><<<N, 256, 0, stream>>>(x, sq, nidx, N);
    build_feat_kernel<<<(M * 32 + 255) / 256, 256, 0, stream>>>(x, 3, 32, nidx, feat, M);
    gemm_wmma<1><<<dim3(Mtiles, 1), 128, 0, stream>>>(feat, W1t, b1, 32, 64, h16, nullptr, 1);
    maxpool_kernel<<<(N * 64 + 255) / 256, 256, 0, stream>>>(h16, x1, Hcat, 0, N);

    // --- EdgeConv 2: kNN in 64-D feature space; K = 128
    sqnorm_kernel<64><<<(N + 255) / 256, 256, 0, stream>>>(x1, sq, N);
    knn_kernel<64><<<N, 256, 0, stream>>>(x1, sq, nidx, N);
    build_feat_kernel<<<(M * 128 + 255) / 256, 256, 0, stream>>>(x1, 64, 128, nidx, feat, M);
    gemm_wmma<1><<<dim3(Mtiles, 1), 128, 0, stream>>>(feat, W2t, b2, 128, 64, h16, nullptr, 1);
    maxpool_kernel<<<(N * 64 + 255) / 256, 256, 0, stream>>>(h16, x2, Hcat, 64, N);

    // --- EdgeConv 3
    sqnorm_kernel<64><<<(N + 255) / 256, 256, 0, stream>>>(x2, sq, N);
    knn_kernel<64><<<N, 256, 0, stream>>>(x2, sq, nidx, N);
    build_feat_kernel<<<(M * 128 + 255) / 256, 256, 0, stream>>>(x2, 64, 128, nidx, feat, M);
    gemm_wmma<1><<<dim3(Mtiles, 1), 128, 0, stream>>>(feat, W3t, b3, 128, 64, h16, nullptr, 1);
    maxpool_kernel<<<(N * 64 + 255) / 256, 256, 0, stream>>>(h16, x3, Hcat, 128, N);

    // --- Classifier MLP (all WMMA; NT = A-fragment reuse per wave)
    gemm_wmma<4><<<dim3(Mr, 4), 128, 0, stream>>>(Hcat, Wlt, bl, 192, 1024, act1, nullptr, 1);
    gemm_wmma<4><<<dim3(Mr, 1), 128, 0, stream>>>(act1, Wm1t, bm1, 1024, 256, act2, nullptr, 1);
    gemm_wmma<2><<<dim3(Mr, 1), 128, 0, stream>>>(act2, Wm2t, bm2, 256, 128, act3, nullptr, 1);
    gemm_wmma<1><<<dim3(Mr, 1), 128, 0, stream>>>(act3, Wot, bo, 128, 50, nullptr, outF, 0);

    lsm_kernel<<<N, 64, 0, stream>>>(outF, 50);
}
